// Decoder_51917564674095
// MI455X (gfx1250) — matrix operand order repaired and checked by reading, not compiled, not run on hardware
//
#include <hip/hip_runtime.h>

// ---------------------------------------------------------------------------
// CDNA5 (gfx1250) decoder step: embed-gather + LSTM cell + flash attention +
// projection.  Memory-bound (~705 MB HBM => ~30us @ 23.3 TB/s).
// GEMMs: v_wmma_f32_16x16x32_bf16 (fp32 accumulate), weights staged through a
// double-buffered LDS tile with a 2-deep software pipeline, split-K over 2
// workgroups per output strip.  Hot-loop barriers use an LDS-only split
// barrier (s_wait_dscnt 0 + s_barrier_signal/wait) so in-flight global loads
// are NOT drained at each sync point, unlike __syncthreads() which emits
// s_wait_loadcnt_dscnt 0x0.
// ---------------------------------------------------------------------------

typedef __attribute__((ext_vector_type(16))) __bf16 v16bf;
typedef __attribute__((ext_vector_type(8)))  __bf16 v8bf;
typedef __attribute__((ext_vector_type(8)))  float  v8f;
typedef __attribute__((ext_vector_type(4)))  float  v4f;

#define BB   128      // batch
#define HH   2048     // hidden
#define SS   512      // source length
#define KA   4096     // K of gates GEMM ( [x|hx] )
#define NC   8        // attention chunks
#define SC   (SS/NC)  // source positions per chunk (64)
#define KT2  64       // k-steps per split-K half (2048/32)

__device__ __forceinline__ float sigmoidf_(float x) {
    return 1.0f / (1.0f + __expf(-x));
}

// LDS-only workgroup barrier: flush this wave's DS ops (so its ds_store is
// visible), then split-barrier.  Deliberately does NOT wait on outstanding
// global loads (private VGPR destinations), keeping the staging/prefetch
// pipeline in flight across the sync.
__device__ __forceinline__ void wg_barrier_lds() {
    asm volatile("s_wait_dscnt 0x0\n\t"
                 "s_barrier_signal -1\n\t"
                 "s_barrier_wait -1" ::: "memory");
}

// ---------------------------------------------------------------------------
// WMMA A-fragment addressing (16x32 bf16, wave32, ISA 7.12.2):
//   lane<16 : m=lane,    elems 0..7 -> K base+0..7,  elems 8..15 -> K base+16..23
//   lane>=16: m=lane-16, elems 0..7 -> K base+8..15, elems 8..15 -> K base+24..31
// Fragment buffer layout: A[((ktile*8 + mtile)*32 + lane)*16 + j]  (bf16)
// ---------------------------------------------------------------------------
__device__ __forceinline__ void frag_coords(int e, int& m, int& k) {
    int j     = e & 15;
    int lane  = (e >> 4) & 31;
    int mtile = (e >> 9) & 7;
    int ktile = e >> 12;
    m = mtile * 16 + (lane & 15);
    int koff = ((lane >> 4) * 8) + j + ((j >= 8) ? 8 : 0);
    k = ktile * 32 + koff;
}

// K0: build A1 = [embed[tw] | hx] as bf16 fragments (128 x 4096)
__global__ __launch_bounds__(256) void pack_A1_kernel(
    const int* __restrict__ tw, const float* __restrict__ embed,
    const float* __restrict__ hx, __bf16* __restrict__ A)
{
    int e = blockIdx.x * 256 + threadIdx.x;   // 2048 blocks -> 524288 elements
    int m, k;
    frag_coords(e, m, k);
    float v = (k < HH) ? embed[(size_t)tw[m] * HH + k]
                       : hx[(size_t)m * HH + (k - HH)];
    A[e] = (__bf16)v;
}

// K4: build A2 = [content | hx_new] as bf16 fragments
__global__ __launch_bounds__(256) void pack_A2_kernel(
    const float* __restrict__ content, const float* __restrict__ hxn,
    __bf16* __restrict__ A)
{
    int e = blockIdx.x * 256 + threadIdx.x;
    int m, k;
    frag_coords(e, m, k);
    float v = (k < HH) ? content[(size_t)m * HH + k]
                       : hxn[(size_t)m * HH + (k - HH)];
    A[e] = (__bf16)v;
}

// ---------------------------------------------------------------------------
// Pipelined cooperative staging of NG 32x16 bf16 B-fragments.
// stage_issue: fire the NT global loads (no wait); stage_commit: cvt + LDS
// store one iteration later, so the HBM latency is covered by a full k-step
// of WMMA/DS work from all waves.
// Fragment layout in LDS (per fragment g):
//   lane<16 : n = row0+g*grs+lane,    elems 0..7 -> K = kkbase + 0..7,  elems 8..15 -> K = kkbase + 16..23
//   lane>=16: n = row0+g*grs+lane-16, elems 0..7 -> K = kkbase + 8..15, elems 8..15 -> K = kkbase + 24..31
// ---------------------------------------------------------------------------
template <int NG>
__device__ __forceinline__ void stage_issue(
    const float* __restrict__ W, int ldk, int row0, int grs,
    int kkbase, int tid, v4f& w0, v4f& w1)
{
    if (NG * 64 >= 256 || tid < NG * 64) {
        const int half = tid & 1;
        const int pair = tid >> 1;
        const int g    = pair >> 5;
        const int lane = pair & 31;
        const int n    = row0 + g * grs + (lane & 15);
        const int kk   = kkbase + (lane >> 4) * 8 + half * 16;
        const v4f* wp  = (const v4f*)(W + (size_t)n * ldk + kk);
        w0 = __builtin_nontemporal_load(wp);
        w1 = __builtin_nontemporal_load(wp + 1);
    }
}

template <int NG>
__device__ __forceinline__ void stage_commit(
    __bf16* __restrict__ sb, int tid, v4f w0, v4f w1)
{
    if (NG * 64 >= 256 || tid < NG * 64) {
        const int half = tid & 1;
        const int pair = tid >> 1;
        const int g    = pair >> 5;
        const int lane = pair & 31;
        v8bf o;
        o[0] = (__bf16)w0.x; o[1] = (__bf16)w0.y; o[2] = (__bf16)w0.z; o[3] = (__bf16)w0.w;
        o[4] = (__bf16)w1.x; o[5] = (__bf16)w1.y; o[6] = (__bf16)w1.z; o[7] = (__bf16)w1.w;
        *(v8bf*)(sb + ((g * 32 + lane) * 16 + half * 8)) = o;
    }
}

// ---------------------------------------------------------------------------
// K1a: partial gates.  Split-K: khalf=0 -> x @ W_ih, khalf=1 -> hx @ W_hh.
// 256 workgroups (2 per 16-wide h strip), 8 waves = 8 M-tiles; each wave
// keeps 4 accumulators (i,f,g,o).  Raw f32 partials to gpart.
// ---------------------------------------------------------------------------
__global__ __launch_bounds__(256) void gates_partial_kernel(
    const __bf16* __restrict__ A1,
    const float* __restrict__ W_ih, const float* __restrict__ W_hh,
    float* __restrict__ gpart)
{
    __shared__ alignas(32) __bf16 sb[2][4 * 32 * 16];   // 8 KB double buffer

    const int tid   = threadIdx.x;
    const int khalf = blockIdx.x & 1;
    const int h0    = (blockIdx.x >> 1) * 16;
    const int lane  = tid & 31;
    const int mtile = tid >> 5;
    const int ln    = lane & 15;
    const int lh    = lane >> 4;
    const int kt0   = khalf * KT2;                      // A-fragment k-tile base

    const float* W = khalf ? W_hh : W_ih;               // [8192][2048]
    float* gout = gpart + (size_t)khalf * BB * (4 * HH);

    v8f acc[4] = {};
    const v16bf* Af = (const v16bf*)A1;

    // pipeline prologue: LDS[0] <- step 0, step 1 loads in flight
    v4f w0, w1;
    stage_issue<4>(W, HH, h0, HH, 0, tid, w0, w1);
    stage_commit<4>(sb[0], tid, w0, w1);
    stage_issue<4>(W, HH, h0, HH, 32, tid, w0, w1);
    v16bf a_cur = Af[((kt0 + 0) * 8 + mtile) * 32 + lane];
    wg_barrier_lds();

    int buf = 0;
    for (int kt = 0; kt < KT2; ++kt) {
        if (kt + 1 < KT2) stage_commit<4>(sb[buf ^ 1], tid, w0, w1);
        if (kt + 2 < KT2) stage_issue<4>(W, HH, h0, HH, (kt + 2) * 32, tid, w0, w1);
        v16bf a_nxt = a_cur;
        if (kt + 1 < KT2) a_nxt = Af[((kt0 + kt + 1) * 8 + mtile) * 32 + lane];
        // hoist all 4 B fragments so the compiler can use incremental dscnt waits
        v16bf bfr[4];
#pragma unroll
        for (int g = 0; g < 4; ++g)
            bfr[g] = *(const v16bf*)(sb[buf] + (g * 32 + lane) * 16);
#pragma unroll
        for (int g = 0; g < 4; ++g)
            acc[g] = __builtin_amdgcn_wmma_f32_16x16x32_bf16(
                false, a_cur, false, bfr[g], (short)0, acc[g], false, false);
        wg_barrier_lds();
        buf ^= 1;
        a_cur = a_nxt;
    }

    // raw partial store (C/D layout: elem r, lane>=16 -> M+8)
#pragma unroll
    for (int g = 0; g < 4; ++g) {
#pragma unroll
        for (int r = 0; r < 8; ++r) {
            const int brow = mtile * 16 + lh * 8 + r;
            gout[(size_t)brow * (4 * HH) + g * HH + h0 + ln] = acc[g][r];
        }
    }
}

// K1b: LSTM epilogue: sum split-K halves + bias, gate nonlinearities, cell.
__global__ __launch_bounds__(256) void lstm_ep_kernel(
    const float* __restrict__ gpart,
    const float* __restrict__ b_ih, const float* __restrict__ b_hh,
    const float* __restrict__ cx, float* __restrict__ hxn)
{
    const int idx = blockIdx.x * 256 + threadIdx.x;   // 1024 blocks -> B*H
    const int b = idx >> 11, h = idx & (HH - 1);
    const float* g0 = gpart + (size_t)b * (4 * HH) + h;
    const float* g1 = g0 + (size_t)BB * (4 * HH);
    float gi = g0[0]      + g1[0]      + b_ih[h]          + b_hh[h];
    float gf = g0[HH]     + g1[HH]     + b_ih[h + HH]     + b_hh[h + HH];
    float gg = g0[2 * HH] + g1[2 * HH] + b_ih[h + 2 * HH] + b_hh[h + 2 * HH];
    float go = g0[3 * HH] + g1[3 * HH] + b_ih[h + 3 * HH] + b_hh[h + 3 * HH];
    float iv = sigmoidf_(gi), fv = sigmoidf_(gf);
    float gv = tanhf(gg),     ov = sigmoidf_(go);
    float c  = fv * cx[idx] + iv * gv;
    hxn[idx] = ov * tanhf(c);
}

// ---------------------------------------------------------------------------
// K2: flash-style attention partials.  grid = (128 b, 8 chunks), 256 threads.
// Each thread owns 8 h-columns; per source position: dot-reduce (shfl+LDS),
// mask, online-softmax rescale, accumulate content in registers.
// ew_hx_list streamed nontemporally; loads for s+1 issued before the
// reduction of step s and kept in flight across the LDS-only barriers.
// ---------------------------------------------------------------------------
__global__ __launch_bounds__(256) void attn_partial_kernel(
    const float* __restrict__ hxn, const float* __restrict__ ew,
    const int* __restrict__ mask,
    float* __restrict__ pc, float* __restrict__ pm, float* __restrict__ pl)
{
    __shared__ float red[8];
    const int b    = blockIdx.x;
    const int c    = blockIdx.y;
    const int tid  = threadIdx.x;
    const int lane = tid & 31;
    const int wid  = tid >> 5;
    const int hb   = tid * 8;

    float h8[8];
    {
        const v4f* hp = (const v4f*)(hxn + (size_t)b * HH + hb);
        v4f a = hp[0], d = hp[1];
        h8[0] = a.x; h8[1] = a.y; h8[2] = a.z; h8[3] = a.w;
        h8[4] = d.x; h8[5] = d.y; h8[6] = d.z; h8[7] = d.w;
    }
    float acc[8] = {0, 0, 0, 0, 0, 0, 0, 0};
    float mrun = -3.402823466e38f, lrun = 0.0f;

    const int s0 = c * SC;
    const v4f* ep0 = (const v4f*)(ew + ((size_t)s0 * BB + b) * HH + hb);
    v4f e0 = __builtin_nontemporal_load(ep0);
    v4f e1 = __builtin_nontemporal_load(ep0 + 1);
    int  mk = mask[s0 * BB + b];

    for (int s = s0; s < s0 + SC; ++s) {
        float er[8] = {e0.x, e0.y, e0.z, e0.w, e1.x, e1.y, e1.z, e1.w};
        const int mcur = mk;
        if (s + 1 < s0 + SC) {                       // prefetch next step
            const v4f* ep = (const v4f*)(ew + ((size_t)(s + 1) * BB + b) * HH + hb);
            e0 = __builtin_nontemporal_load(ep);
            e1 = __builtin_nontemporal_load(ep + 1);
            mk = mask[(s + 1) * BB + b];
        }

        float part = 0.0f;
#pragma unroll
        for (int j = 0; j < 8; ++j) part += er[j] * h8[j];
#pragma unroll
        for (int off = 16; off > 0; off >>= 1) part += __shfl_xor(part, off, 32);
        if (lane == 0) red[wid] = part;
        wg_barrier_lds();                            // prefetch stays in flight
        float score = red[0] + red[1] + red[2] + red[3] +
                      red[4] + red[5] + red[6] + red[7];
        wg_barrier_lds();

        if (mcur == 0) score = -1e9f;               // NEG_INF (finite, like ref)

        float nm = fmaxf(mrun, score);
        float sc = __expf(mrun - nm);               // first iter: exp(-huge)=0
        float p  = __expf(score - nm);
        lrun = lrun * sc + p;
#pragma unroll
        for (int j = 0; j < 8; ++j) acc[j] = acc[j] * sc + p * er[j];
        mrun = nm;
    }

    float* pcp = pc + ((size_t)b * NC + c) * HH + hb;
#pragma unroll
    for (int j = 0; j < 8; ++j) pcp[j] = acc[j];
    if (tid == 0) { pm[b * NC + c] = mrun; pl[b * NC + c] = lrun; }
}

// K3: combine chunk partials -> content[b][h]
__global__ __launch_bounds__(256) void attn_combine_kernel(
    const float* __restrict__ pc, const float* __restrict__ pm,
    const float* __restrict__ pl, float* __restrict__ content)
{
    const int b  = blockIdx.x;
    const int hb = threadIdx.x * 8;

    float gm = -3.402823466e38f;
#pragma unroll
    for (int c = 0; c < NC; ++c) gm = fmaxf(gm, pm[b * NC + c]);
    float w[NC];
    float tot = 0.0f;
#pragma unroll
    for (int c = 0; c < NC; ++c) {
        w[c] = __expf(pm[b * NC + c] - gm);
        tot += pl[b * NC + c] * w[c];
    }
    const float inv = 1.0f / tot;

    float acc[8] = {0, 0, 0, 0, 0, 0, 0, 0};
#pragma unroll
    for (int c = 0; c < NC; ++c) {
        const float* p = pc + ((size_t)b * NC + c) * HH + hb;
#pragma unroll
        for (int j = 0; j < 8; ++j) acc[j] += w[c] * p[j];
    }
    float* op = content + (size_t)b * HH + hb;
#pragma unroll
    for (int j = 0; j < 8; ++j) op[j] = acc[j] * inv;
}

// ---------------------------------------------------------------------------
// K5a: partial projection.  Split-K: khalf=0 -> content cols, khalf=1 -> hx
// cols of attn_W.  256 workgroups (2 per 16-wide n strip), raw f32 partials.
// ---------------------------------------------------------------------------
__global__ __launch_bounds__(256) void proj_partial_kernel(
    const __bf16* __restrict__ A2, const float* __restrict__ attn_W,
    float* __restrict__ ppart)
{
    __shared__ alignas(32) __bf16 sb[2][32 * 16];   // 2 KB double buffer

    const int tid   = threadIdx.x;
    const int khalf = blockIdx.x & 1;
    const int n0    = (blockIdx.x >> 1) * 16;
    const int lane  = tid & 31;
    const int mtile = tid >> 5;
    const int ln    = lane & 15;
    const int lh    = lane >> 4;
    const int kt0   = khalf * KT2;
    const int kb    = khalf * HH;                    // column base in attn_W

    float* pout = ppart + (size_t)khalf * BB * HH;

    v8f acc = {};
    const v16bf* Af = (const v16bf*)A2;

    v4f w0, w1;
    stage_issue<1>(attn_W, KA, n0, 0, kb, tid, w0, w1);
    stage_commit<1>(sb[0], tid, w0, w1);
    stage_issue<1>(attn_W, KA, n0, 0, kb + 32, tid, w0, w1);
    v16bf a_cur = Af[((kt0 + 0) * 8 + mtile) * 32 + lane];
    wg_barrier_lds();

    int buf = 0;
    for (int kt = 0; kt < KT2; ++kt) {
        if (kt + 1 < KT2) stage_commit<1>(sb[buf ^ 1], tid, w0, w1);
        if (kt + 2 < KT2) stage_issue<1>(attn_W, KA, n0, 0, kb + (kt + 2) * 32, tid, w0, w1);
        v16bf a_nxt = a_cur;
        if (kt + 1 < KT2) a_nxt = Af[((kt0 + kt + 1) * 8 + mtile) * 32 + lane];
        v16bf b = *(const v16bf*)(sb[buf] + lane * 16);
        acc = __builtin_amdgcn_wmma_f32_16x16x32_bf16(
            false, a_cur, false, b, (short)0, acc, false, false);
        wg_barrier_lds();
        buf ^= 1;
        a_cur = a_nxt;
    }

#pragma unroll
    for (int r = 0; r < 8; ++r) {
        const int brow = mtile * 16 + lh * 8 + r;
        pout[(size_t)brow * HH + n0 + ln] = acc[r];
    }
}

// K5b: projection epilogue: sum halves + bias + tanh -> hx_attention.
__global__ __launch_bounds__(256) void proj_ep_kernel(
    const float* __restrict__ ppart, const float* __restrict__ attn_b,
    float* __restrict__ out)
{
    const int idx = blockIdx.x * 256 + threadIdx.x;   // 1024 blocks -> B*H
    const int n = idx & (HH - 1);
    out[idx] = tanhf(ppart[idx] + ppart[idx + (size_t)BB * HH] + attn_b[n]);
}

// ---------------------------------------------------------------------------
extern "C" void kernel_launch(void* const* d_in, const int* in_sizes, int n_in,
                              void* d_out, int out_size, void* d_ws, size_t ws_size,
                              hipStream_t stream)
{
    const int*   tw     = (const int*)  d_in[0];
    const float* hx     = (const float*)d_in[1];
    const float* cx     = (const float*)d_in[2];
    const float* ew     = (const float*)d_in[3];
    const int*   mask   = (const int*)  d_in[4];
    const float* embed  = (const float*)d_in[5];
    const float* W_ih   = (const float*)d_in[6];
    const float* W_hh   = (const float*)d_in[7];
    const float* b_ih   = (const float*)d_in[8];
    const float* b_hh   = (const float*)d_in[9];
    const float* attn_W = (const float*)d_in[10];
    const float* attn_b = (const float*)d_in[11];
    float* out = (float*)d_out;

    // workspace partition (all sizes 16B-aligned), ~22.1 MB total
    char* p = (char*)d_ws;
    __bf16* A1      = (__bf16*)p;  p += (size_t)BB * KA * 2;           // 1 MB
    __bf16* A2      = (__bf16*)p;  p += (size_t)BB * KA * 2;           // 1 MB
    float*  hxn     = (float*)p;   p += (size_t)BB * HH * 4;           // 1 MB
    float*  content = (float*)p;   p += (size_t)BB * HH * 4;           // 1 MB
    float*  pc      = (float*)p;   p += (size_t)BB * NC * HH * 4;      // 8 MB
    float*  pm      = (float*)p;   p += (size_t)BB * NC * 4;
    float*  pl      = (float*)p;   p += (size_t)BB * NC * 4;
    float*  gpart   = (float*)p;   p += (size_t)2 * BB * 4 * HH * 4;   // 8 MB
    float*  ppart   = (float*)p;   p += (size_t)2 * BB * HH * 4;       // 2 MB

    pack_A1_kernel       <<<2048, 256, 0, stream>>>(tw, embed, hx, A1);
    gates_partial_kernel <<<2 * (HH / 16), 256, 0, stream>>>(A1, W_ih, W_hh, gpart);
    lstm_ep_kernel       <<<(BB * HH) / 256, 256, 0, stream>>>(gpart, b_ih, b_hh, cx, hxn);
    dim3 g2(BB, NC);
    attn_partial_kernel  <<<g2, 256, 0, stream>>>(hxn, ew, mask, pc, pm, pl);
    attn_combine_kernel  <<<BB, 256, 0, stream>>>(pc, pm, pl, content);
    pack_A2_kernel       <<<2048, 256, 0, stream>>>(content, hxn, A2);
    proj_partial_kernel  <<<2 * (HH / 16), 256, 0, stream>>>(A2, attn_W, ppart);
    proj_ep_kernel       <<<(BB * HH) / 256, 256, 0, stream>>>(ppart, attn_b, out);
}
